// DNC_34557306864265
// MI455X (gfx1250) — matrix operand order repaired and checked by reading, not compiled, not run on hardware
//
#include <hip/hip_runtime.h>
#include <hip/hip_bf16.h>
#include <math.h>

typedef __attribute__((ext_vector_type(16))) _Float16 v16h;
typedef __attribute__((ext_vector_type(8)))  _Float16 v8h;
typedef __attribute__((ext_vector_type(8)))  float    v8f;
typedef __attribute__((ext_vector_type(4)))  float    v4f;

#define DEVINL __device__ __forceinline__

#define Bb   64
#define Nn   1024
#define Wd   64
#define Rr   4
#define Cc   512
#define IFP  480     // padded iface pitch
#define IFN  471     // valid iface columns

DEVINL float sigf(float x)      { return 1.f / (1.f + expf(-x)); }
DEVINL float softplusf_(float x){ return fmaxf(x, 0.f) + log1pf(expf(-fabsf(x))); }

// ---------------------------------------------------------------- reductions
DEVINL float blockReduceMax256(float v, float* red) {
    int t = threadIdx.x;
    red[t] = v; __syncthreads();
    for (int s = 128; s > 0; s >>= 1) {
        if (t < s) red[t] = fmaxf(red[t], red[t + s]);
        __syncthreads();
    }
    float r = red[0]; __syncthreads();
    return r;
}
DEVINL float blockReduceSum256(float v, float* red) {
    int t = threadIdx.x;
    red[t] = v; __syncthreads();
    for (int s = 128; s > 0; s >>= 1) {
        if (t < s) red[t] = red[t] + red[t + s];
        __syncthreads();
    }
    float r = red[0]; __syncthreads();
    return r;
}

// ---------------------------------------------------------------- converts
__global__ void conv_f32_f16(const float* __restrict__ src, _Float16* __restrict__ dst, int n) {
    int i = blockIdx.x * 256 + threadIdx.x;
    if (i < n) dst[i] = (_Float16)src[i];
}

__global__ void conv_wif(const float* __restrict__ src, _Float16* __restrict__ dst) {
    int i = blockIdx.x * 256 + threadIdx.x;          // 480*512
    if (i >= IFP * Cc) return;
    int row = i >> 9, k = i & 511;
    dst[i] = (row < IFN) ? (_Float16)src[row * Cc + k] : (_Float16)0.f;
}

__global__ void build_afull(const float* __restrict__ x, const float* __restrict__ rd,
                            const float* __restrict__ h, _Float16* __restrict__ A) {
    int i = blockIdx.x * 256 + threadIdx.x;          // 64*1024
    if (i >= Bb * 1024) return;
    int b = i >> 10, k = i & 1023;
    float v = (k < 256) ? x[b * 256 + k]
            : (k < 512) ? rd[b * 256 + (k - 256)]
                        : h[b * Cc + (k - 512)];
    A[i] = (_Float16)v;
}

__global__ void zero_f32(float* __restrict__ p, int n) {
    int i = blockIdx.x * 256 + threadIdx.x;
    if (i < n) p[i] = 0.f;
}

// ---------------------------------------------------------------- WMMA GEMM
// C[64 x nOut] = A[64 x k1]*W1^T (+ A[:, k1:k1+k2]*W2^T) + b1 (+ b2)
// A: f16 row-major pitch lda.  W: f16 row-major [nOut x K] (so W^T has K rows).
__global__ __launch_bounds__(256) void wmma_gemm(
    const _Float16* __restrict__ A, int lda,
    const _Float16* __restrict__ W1, int k1,
    const _Float16* __restrict__ W2, int k2,
    const float* __restrict__ b1, const float* __restrict__ b2,
    float* __restrict__ C, int ldc, int nOut, int nValid)
{
    int ntiles = (nOut + 15) >> 4;
    int wid = blockIdx.x * (blockDim.x >> 5) + (threadIdx.x >> 5);
    if (wid >= 4 * ntiles) return;
    int mt = wid / ntiles, nt = wid % ntiles;

    int lane = threadIdx.x & 31;
    int half = lane >> 4;          // K-half selector
    int lrow = lane & 15;          // A row / B col within tile

    v8f c = {};

    // phase 1: A columns [0, k1), weights W1
    {
        const _Float16* ap = A  + (size_t)(mt * 16 + lrow) * lda + half * 16;
        const _Float16* bp = W1 + (size_t)(nt * 16 + lrow) * k1  + half * 16;
        for (int k0 = 0; k0 < k1; k0 += 32) {
            v8h alo = *(const v8h*)(ap + k0);
            v8h ahi = *(const v8h*)(ap + k0 + 8);
            v16h a = __builtin_shufflevector(alo, ahi, 0,1,2,3,4,5,6,7,8,9,10,11,12,13,14,15);
            v16h bmat = *(const v16h*)(bp + k0);
            c = __builtin_amdgcn_wmma_f32_16x16x32_f16(false, a, false, bmat,
                                                       (short)0, c, false, false);
        }
    }
    // phase 2: A columns [k1, k1+k2), weights W2
    if (W2) {
        const _Float16* ap = A  + (size_t)(mt * 16 + lrow) * lda + k1 + half * 16;
        const _Float16* bp = W2 + (size_t)(nt * 16 + lrow) * k2  + half * 16;
        for (int k0 = 0; k0 < k2; k0 += 32) {
            v8h alo = *(const v8h*)(ap + k0);
            v8h ahi = *(const v8h*)(ap + k0 + 8);
            v16h a = __builtin_shufflevector(alo, ahi, 0,1,2,3,4,5,6,7,8,9,10,11,12,13,14,15);
            v16h bmat = *(const v16h*)(bp + k0);
            c = __builtin_amdgcn_wmma_f32_16x16x32_f16(false, a, false, bmat,
                                                       (short)0, c, false, false);
        }
    }

    int col = nt * 16 + lrow;
    if (col < nValid) {
        float bias = (b1 ? b1[col] : 0.f) + (b2 ? b2[col] : 0.f);
#pragma unroll
        for (int v = 0; v < 8; ++v) {
            int m = mt * 16 + half * 8 + v;
            C[(size_t)m * ldc + col] = c[v] + bias;
        }
    }
}

// ---------------------------------------------------------------- LSTM
__global__ void lstm_kernel(const float* __restrict__ gates, const float* __restrict__ c_in,
                            float* __restrict__ c_new, float* __restrict__ ctrl,
                            _Float16* __restrict__ Aout) {
    int i = blockIdx.x * 256 + threadIdx.x;          // 64*512
    if (i >= Bb * Cc) return;
    int b = i >> 9, j = i & 511;
    const float* g = gates + (size_t)b * (4 * Cc);
    float ig = sigf(g[j]);
    float fg = sigf(g[Cc + j]);
    float gg = tanhf(g[2 * Cc + j]);
    float og = sigf(g[3 * Cc + j]);
    float cn = fg * c_in[i] + ig * gg;
    float ct = og * tanhf(cn);
    c_new[i] = cn;
    ctrl[i]  = ct;
    Aout[(size_t)b * 768 + j] = (_Float16)ct;
}

// ---------------------------------------------------------------- usage / retention
__global__ void usage_kernel(const float* __restrict__ iface, const float* __restrict__ r_w,
                             const float* __restrict__ w_w, const float* __restrict__ usage,
                             float* __restrict__ usage_u) {
    int b = blockIdx.x, t = threadIdx.x;
    const float* ifb = iface + b * IFP;
    float sf[Rr];
#pragma unroll
    for (int r = 0; r < Rr; ++r) sf[r] = sigf(ifb[453 + r]);
#pragma unroll
    for (int q = 0; q < 4; ++q) {
        int n = t + q * 256;
        float ret = 1.f;
#pragma unroll
        for (int r = 0; r < Rr; ++r)
            ret *= 1.f - sf[r] * r_w[((size_t)b * Rr + r) * Nn + n];
        float u = usage[b * Nn + n], w = w_w[b * Nn + n];
        usage_u[b * Nn + n] = (u + w - u * w) * ret;
    }
}

// ---------------------------------------------------------------- sort + allocation
__global__ __launch_bounds__(512) void sort_alloc_kernel(const float* __restrict__ usage_u,
                                                         float* __restrict__ alloc) {
    __shared__ float v[Nn];
    __shared__ int   ix[Nn];
    __shared__ float cpA[Nn];
    __shared__ float cpB[Nn];
    int b = blockIdx.x, t = threadIdx.x;
    for (int i = t; i < Nn; i += 512) { v[i] = usage_u[b * Nn + i]; ix[i] = i; }
    __syncthreads();
    // bitonic ascending sort (index tie-break matches stable argsort)
    for (int k = 2; k <= Nn; k <<= 1) {
        for (int j = k >> 1; j > 0; j >>= 1) {
            for (int i = t; i < Nn; i += 512) {
                int p = i ^ j;
                if (p > i) {
                    bool up = ((i & k) == 0);
                    float a = v[i], bb = v[p];
                    int ia = ix[i], ib = ix[p];
                    bool sw = up ? (a > bb || (a == bb && ia > ib))
                                 : (a < bb || (a == bb && ia < ib));
                    if (sw) { v[i] = bb; v[p] = a; ix[i] = ib; ix[p] = ia; }
                }
            }
            __syncthreads();
        }
    }
    // inclusive cumprod (Hillis-Steele, ping-pong)
    for (int i = t; i < Nn; i += 512) cpA[i] = v[i];
    __syncthreads();
    float* src = cpA; float* dst = cpB;
    for (int off = 1; off < Nn; off <<= 1) {
        for (int i = t; i < Nn; i += 512)
            dst[i] = (i >= off) ? src[i] * src[i - off] : src[i];
        __syncthreads();
        float* tmp = src; src = dst; dst = tmp;
    }
    for (int i = t; i < Nn; i += 512)
        alloc[b * Nn + ix[i]] = (1.f - v[i]) * src[i];
}

// ---------------------------------------------------------------- write weighting + precedence
__global__ __launch_bounds__(256) void write_weight_kernel(
    const float* __restrict__ iface, const float* __restrict__ memory,
    const float* __restrict__ alloc, const float* __restrict__ precedence,
    float* __restrict__ ww_out, float* __restrict__ prec_out)
{
    __shared__ float wk[Wd];
    __shared__ float red[256];
    int b = blockIdx.x, t = threadIdx.x;
    const float* ifb = iface + b * IFP;
    if (t < Wd) wk[t] = ifb[260 + t];
    __syncthreads();
    float ss = 0.f;
#pragma unroll
    for (int w = 0; w < Wd; ++w) ss += wk[w] * wk[w];
    float invk = 1.f / (sqrtf(ss) + 1e-8f);
    float beta = 1.f + softplusf_(ifb[324]);
    float ag = sigf(ifb[457]);
    float wg = sigf(ifb[458]);

    float sc[4];
#pragma unroll
    for (int q = 0; q < 4; ++q) {
        int n = t + q * 256;
        const float* mrow = memory + ((size_t)(b * Nn + n)) * Wd;
        float d = 0.f, m2 = 0.f;
#pragma unroll
        for (int w = 0; w < Wd; ++w) { float mv = mrow[w]; d += mv * wk[w]; m2 += mv * mv; }
        sc[q] = d * invk * beta / (sqrtf(m2) + 1e-8f);
    }
    float mx = blockReduceMax256(fmaxf(fmaxf(sc[0], sc[1]), fmaxf(sc[2], sc[3])), red);
    float ex[4]; float le = 0.f;
#pragma unroll
    for (int q = 0; q < 4; ++q) { ex[q] = expf(sc[q] - mx); le += ex[q]; }
    float sm = blockReduceSum256(le, red);

    float wwq[4]; float lw = 0.f;
#pragma unroll
    for (int q = 0; q < 4; ++q) {
        int n = t + q * 256;
        float wc = ex[q] / sm;
        float w = wg * (ag * alloc[b * Nn + n] + (1.f - ag) * wc);
        wwq[q] = w; lw += w;
        ww_out[b * Nn + n] = w;
    }
    float S = blockReduceSum256(lw, red);
#pragma unroll
    for (int q = 0; q < 4; ++q) {
        int n = t + q * 256;
        prec_out[b * Nn + n] = (1.f - S) * precedence[b * Nn + n] + wwq[q];
    }
}

// ---------------------------------------------------------------- memory update
__global__ void memnew_kernel(const float* __restrict__ memory, const float* __restrict__ ww,
                              const float* __restrict__ iface, float* __restrict__ mem_out) {
    int i = blockIdx.x * 256 + threadIdx.x;          // 64*1024*64
    if (i >= Bb * Nn * Wd) return;
    int w = i & 63, n = (i >> 6) & 1023, b = i >> 16;
    const float* ifb = iface + b * IFP;
    float ev = sigf(ifb[325 + w]);
    float wv = ifb[389 + w];
    float wn = ww[b * Nn + n];
    mem_out[i] = memory[i] * (1.f - wn * ev) + wn * wv;
}

// ---------------------------------------------------------------- fused link update + fwd + bwd
// block = (b, 128x128 tile).  fwd[b,r,n]=sum_m L'[n,m]*rw[r,m],  bwd[b,r,n]=sum_m L'[m,n]*rw[r,m]
// link is a 536MB one-shot stream: use non-temporal load/store (TH_NT) so the
// 192MB L2 keeps the reused working set (mem_new, weights, per-batch vectors).
__global__ __launch_bounds__(256) void link_fwd_bwd_kernel(
    const float* __restrict__ link, const float* __restrict__ ww,
    const float* __restrict__ precedence, const float* __restrict__ r_w,
    float* __restrict__ link_out, float* __restrict__ fwd, float* __restrict__ bwd)
{
    int bi = blockIdx.x;
    int b = bi >> 6, tile = bi & 63;
    int n0 = (tile >> 3) << 7, m0 = (tile & 7) << 7;
    int t = threadIdx.x, tx = t & 31, ty = t >> 5;

    __shared__ float wwn_s[128];
    __shared__ float rwn_s[Rr][128];
    __shared__ float bwd_s[Rr][128];

    if (t < 128) {
        wwn_s[t] = ww[b * Nn + n0 + t];
#pragma unroll
        for (int r = 0; r < Rr; ++r)
            rwn_s[r][t] = r_w[((size_t)b * Rr + r) * Nn + n0 + t];
    }
    for (int i = t; i < Rr * 128; i += 256) ((float*)bwd_s)[i] = 0.f;
    __syncthreads();

    int mbase = m0 + tx * 4;
    v4f wv4 = *(const v4f*)(ww + b * Nn + mbase);
    v4f pv4 = *(const v4f*)(precedence + b * Nn + mbase);
    float wwm[4] = {wv4[0], wv4[1], wv4[2], wv4[3]};
    float prm[4] = {pv4[0], pv4[1], pv4[2], pv4[3]};
    float rwm[Rr][4];
#pragma unroll
    for (int r = 0; r < Rr; ++r) {
        v4f rv = *(const v4f*)(r_w + ((size_t)b * Rr + r) * Nn + mbase);
        rwm[r][0] = rv[0]; rwm[r][1] = rv[1]; rwm[r][2] = rv[2]; rwm[r][3] = rv[3];
    }
    float bacc[Rr][4] = {};

    for (int it = 0; it < 16; ++it) {
        int nl = ty + (it << 3);
        int n = n0 + nl;
        float wn = wwn_s[nl];
        float rn[Rr];
#pragma unroll
        for (int r = 0; r < Rr; ++r) rn[r] = rwn_s[r][nl];

        size_t rowoff = ((size_t)(b * Nn + n)) * Nn + mbase;
        v4f l4 = __builtin_nontemporal_load((const v4f*)(link + rowoff));
        float lv[4] = {l4[0], l4[1], l4[2], l4[3]};
        float fp[Rr] = {0.f, 0.f, 0.f, 0.f};
#pragma unroll
        for (int j = 0; j < 4; ++j) {
            int m = mbase + j;
            float val = (1.f - wn - wwm[j]) * lv[j] + wn * prm[j];
            if (m == n) val = 0.f;
            lv[j] = val;
#pragma unroll
            for (int r = 0; r < Rr; ++r) {
                fp[r]      += val * rwm[r][j];
                bacc[r][j] += val * rn[r];
            }
        }
        v4f o4 = {lv[0], lv[1], lv[2], lv[3]};
        __builtin_nontemporal_store(o4, (v4f*)(link_out + rowoff));

        // wave32 reduction of fwd partials across the row
#pragma unroll
        for (int r = 0; r < Rr; ++r) {
            float s = fp[r];
            for (int o = 16; o > 0; o >>= 1) s += __shfl_xor(s, o, 32);
            if (tx == 0) atomicAdd(&fwd[((size_t)b * Rr + r) * Nn + n], s);
        }
    }
    __syncthreads();
#pragma unroll
    for (int r = 0; r < Rr; ++r)
#pragma unroll
        for (int j = 0; j < 4; ++j)
            atomicAdd(&bwd_s[r][tx * 4 + j], bacc[r][j]);
    __syncthreads();
    for (int i = t; i < Rr * 128; i += 256) {
        int r = i >> 7, ml = i & 127;
        atomicAdd(&bwd[((size_t)b * Rr + r) * Nn + m0 + ml], bwd_s[r][ml]);
    }
}

// ---------------------------------------------------------------- read weighting
__global__ __launch_bounds__(256) void read_weight_kernel(
    const float* __restrict__ iface, const float* __restrict__ mem_new,
    const float* __restrict__ fwd, const float* __restrict__ bwd,
    float* __restrict__ rw_out)
{
    __shared__ float rk[Rr][Wd];
    __shared__ float invn_s[Rr];
    __shared__ float beta_s[Rr];
    __shared__ float mode_s[Rr][3];
    __shared__ float red[256];
    int b = blockIdx.x, t = threadIdx.x;
    const float* ifb = iface + b * IFP;

    rk[t >> 6][t & 63] = ifb[t];                       // r_keys: iface[0:256]
    __syncthreads();
    if (t < Rr) {
        float ss = 0.f;
        for (int w = 0; w < Wd; ++w) ss += rk[t][w] * rk[t][w];
        invn_s[t] = 1.f / (sqrtf(ss) + 1e-8f);
        beta_s[t] = 1.f + softplusf_(ifb[256 + t]);
        float e0 = ifb[459 + t * 3], e1 = ifb[460 + t * 3], e2 = ifb[461 + t * 3];
        float mm = fmaxf(e0, fmaxf(e1, e2));
        float x0 = expf(e0 - mm), x1 = expf(e1 - mm), x2 = expf(e2 - mm);
        float s = x0 + x1 + x2;
        mode_s[t][0] = x0 / s; mode_s[t][1] = x1 / s; mode_s[t][2] = x2 / s;
    }
    __syncthreads();
    rk[t >> 6][t & 63] *= invn_s[t >> 6];
    __syncthreads();

    float sc[4][Rr];
#pragma unroll
    for (int q = 0; q < 4; ++q) {
        int n = t + q * 256;
        const float* mrow = mem_new + ((size_t)(b * Nn + n)) * Wd;
        float m2 = 0.f, d[Rr] = {0.f, 0.f, 0.f, 0.f};
#pragma unroll
        for (int w = 0; w < Wd; ++w) {
            float mv = mrow[w];
            m2 += mv * mv;
#pragma unroll
            for (int r = 0; r < Rr; ++r) d[r] += mv * rk[r][w];
        }
        float inv = 1.f / (sqrtf(m2) + 1e-8f);
#pragma unroll
        for (int r = 0; r < Rr; ++r) sc[q][r] = d[r] * inv * beta_s[r];
    }

    for (int r = 0; r < Rr; ++r) {
        float lm = fmaxf(fmaxf(sc[0][r], sc[1][r]), fmaxf(sc[2][r], sc[3][r]));
        float mx = blockReduceMax256(lm, red);
        float ex[4]; float le = 0.f;
#pragma unroll
        for (int q = 0; q < 4; ++q) { ex[q] = expf(sc[q][r] - mx); le += ex[q]; }
        float sm = blockReduceSum256(le, red);
#pragma unroll
        for (int q = 0; q < 4; ++q) {
            int n = t + q * 256;
            float rc = ex[q] / sm;
            size_t idx = ((size_t)b * Rr + r) * Nn + n;
            rw_out[idx] = mode_s[r][0] * bwd[idx] + mode_s[r][1] * rc + mode_s[r][2] * fwd[idx];
        }
    }
}

// ---------------------------------------------------------------- reads = rw_new @ mem_new
__global__ void reads_kernel(const float* __restrict__ rw_new, const float* __restrict__ mem_new,
                             float* __restrict__ reads_out, _Float16* __restrict__ Aout) {
    int b = blockIdx.x, t = threadIdx.x;              // t = r*64 + w
    int r = t >> 6, w = t & 63;
    const float* rwp = rw_new + ((size_t)b * Rr + r) * Nn;
    const float* mp  = mem_new + (size_t)b * Nn * Wd + w;
    float acc = 0.f;
    for (int n = 0; n < Nn; ++n) acc += rwp[n] * mp[(size_t)n * Wd];
    reads_out[b * 256 + t] = acc;
    Aout[(size_t)b * 768 + 512 + t] = (_Float16)acc;
}

// ---------------------------------------------------------------- host launcher
extern "C" void kernel_launch(void* const* d_in, const int* in_sizes, int n_in,
                              void* d_out, int out_size, void* d_ws, size_t ws_size,
                              hipStream_t stream) {
    const float* x     = (const float*)d_in[0];
    const float* rd    = (const float*)d_in[1];
    const float* h     = (const float*)d_in[2];
    const float* c_in  = (const float*)d_in[3];
    const float* mem   = (const float*)d_in[4];
    const float* link  = (const float*)d_in[5];
    const float* rwt   = (const float*)d_in[6];
    const float* wwt   = (const float*)d_in[7];
    const float* usg   = (const float*)d_in[8];
    const float* prc   = (const float*)d_in[9];
    const float* W_ih  = (const float*)d_in[10];
    const float* W_hh  = (const float*)d_in[11];
    const float* b_ih  = (const float*)d_in[12];
    const float* b_hh  = (const float*)d_in[13];
    const float* W_if  = (const float*)d_in[14];
    const float* b_if  = (const float*)d_in[15];
    const float* W_out = (const float*)d_in[16];
    const float* b_out = (const float*)d_in[17];

    char* ws = (char*)d_ws;
    _Float16* Wih_h  = (_Float16*)(ws + 0);
    _Float16* Whh_h  = (_Float16*)(ws + 2097152);
    _Float16* Wif_h  = (_Float16*)(ws + 4194304);
    _Float16* Wout_h = (_Float16*)(ws + 4685824);
    _Float16* Afull  = (_Float16*)(ws + 5079040);
    _Float16* Aout   = (_Float16*)(ws + 5210112);
    float* gates = (float*)(ws + 5308416);
    float* iface = (float*)(ws + 5832704);
    float* alloc = (float*)(ws + 5955584);
    float* fwd   = (float*)(ws + 6217728);
    float* bwd   = (float*)(ws + 7266304);

    float* dout    = (float*)d_out;
    float* o_out   = dout;
    float* o_reads = dout + 16384;
    float* o_mem   = dout + 32768;
    float* o_link  = dout + 4227072;
    float* o_rw    = dout + 71335936;
    float* o_ww    = dout + 71598080;
    float* o_usage = dout + 71663616;
    float* o_prec  = dout + 71729152;
    float* o_ctrl  = dout + 71794688;
    float* o_cnew  = dout + 71827456;

    // precision conversions + input assembly
    conv_f32_f16<<<4096, 256, 0, stream>>>(W_ih, Wih_h, 2048 * 512);
    conv_f32_f16<<<4096, 256, 0, stream>>>(W_hh, Whh_h, 2048 * 512);
    conv_wif<<<960, 256, 0, stream>>>(W_if, Wif_h);
    conv_f32_f16<<<768, 256, 0, stream>>>(W_out, Wout_h, 256 * 768);
    build_afull<<<256, 256, 0, stream>>>(x, rd, h, Afull);
    zero_f32<<<2048, 256, 0, stream>>>(fwd, 2 * Bb * Rr * Nn);   // fwd+bwd contiguous

    // gates = [x|read|h] @ [W_ih|W_hh]^T + b_ih + b_hh          (WMMA)
    wmma_gemm<<<64, 256, 0, stream>>>(Afull, 1024, Wih_h, 512, Whh_h, 512,
                                      b_ih, b_hh, gates, 2048, 2048, 2048);
    lstm_kernel<<<128, 256, 0, stream>>>(gates, c_in, o_cnew, o_ctrl, Aout);

    // iface = ctrl @ W_if^T + b_if                              (WMMA)
    wmma_gemm<<<15, 256, 0, stream>>>(Aout, 768, Wif_h, 512, nullptr, 0,
                                      b_if, nullptr, iface, IFP, IFP, IFN);

    usage_kernel<<<64, 256, 0, stream>>>(iface, rwt, wwt, usg, o_usage);
    sort_alloc_kernel<<<64, 512, 0, stream>>>(o_usage, alloc);
    write_weight_kernel<<<64, 256, 0, stream>>>(iface, mem, alloc, prc, o_ww, o_prec);
    memnew_kernel<<<16384, 256, 0, stream>>>(mem, o_ww, iface, o_mem);

    // fused: link_new + fwd + bwd (single NT pass over the 268MB link matrix)
    link_fwd_bwd_kernel<<<4096, 256, 0, stream>>>(link, o_ww, prc, rwt, o_link, fwd, bwd);

    read_weight_kernel<<<64, 256, 0, stream>>>(iface, o_mem, fwd, bwd, o_rw);
    reads_kernel<<<64, 256, 0, stream>>>(o_rw, o_mem, o_reads, Aout);

    // out = [ctrl|reads] @ W_out^T + b_out                      (WMMA)
    wmma_gemm<<<8, 256, 0, stream>>>(Aout, 768, Wout_h, 768, nullptr, 0,
                                     b_out, nullptr, o_out, 256, 256, 256);
}
